// MultiHeadAttention_12945031430784
// MI455X (gfx1250) — matrix operand order repaired and checked by reading, not compiled, not run on hardware
//
#include <hip/hip_runtime.h>
#include <hip/hip_bf16.h>
#include <stdint.h>

#define NHEAD 8
#define CCH   512
#define HWTOK 1024
#define NBAT  8
#define DH    64
#define LN_EPS 1e-6f

typedef __attribute__((ext_vector_type(16))) __bf16 v16bf;
typedef __attribute__((ext_vector_type(8)))  float  v8f;

union BFrag { uint32_t u[8]; v16bf v; };
union FAcc  { float f[8]; v8f v; };

__device__ __forceinline__ uint16_t f2bf(float f) {
  union { float f; uint32_t u; } x; x.f = f;
  uint32_t r = x.u + 0x7FFFu + ((x.u >> 16) & 1u);
  return (uint16_t)(r >> 16);
}
__device__ __forceinline__ uint32_t pk2bf(float a, float b) {
  return (uint32_t)f2bf(a) | ((uint32_t)f2bf(b) << 16);
}
__device__ __forceinline__ float bf2f(uint16_t h) {
  union { uint32_t u; float f; } x; x.u = ((uint32_t)h) << 16; return x.f;
}

// async global->LDS b128 copy (per-lane, ASYNCcnt-tracked)
__device__ __forceinline__ void async_copy_b128(void* lds_dst, const void* gsrc) {
  uint32_t l = (uint32_t)(uintptr_t)lds_dst;     // low 32 bits = LDS offset
  asm volatile("global_load_async_to_lds_b128 %0, %1, off"
               :: "v"(l), "v"(gsrc) : "memory");
}
__device__ __forceinline__ void async_wait0() {
  asm volatile("s_wait_asynccnt 0x0" ::: "memory");
}

// A-matrix 16x32 bf16 fragment (M on lanes, K across regs; half-lane K split)
__device__ __forceinline__ BFrag load_a_bf16(const uint16_t* base, int ld,
                                             int row0, int k0, int lane) {
  BFrag f;
  int r  = lane & 15;
  int kh = (lane >> 4) << 3;
  const uint16_t* p = base + (size_t)(row0 + r) * ld + k0 + kh;
  uint4 a = *(const uint4*)p;
  uint4 b = *(const uint4*)(p + 16);
  f.u[0]=a.x; f.u[1]=a.y; f.u[2]=a.z; f.u[3]=a.w;
  f.u[4]=b.x; f.u[5]=b.y; f.u[6]=b.z; f.u[7]=b.w;
  return f;
}

// B-matrix 32x16 bf16 fragment (N on lanes, K packed 2 per reg; half-lane K split)
__device__ __forceinline__ BFrag load_b_bf16(const uint16_t* base, int ld,
                                             int k0, int n0, int lane) {
  BFrag f;
  const uint16_t* p = base + (size_t)(k0 + ((lane >> 4) << 3)) * ld + n0 + (lane & 15);
  for (int i = 0; i < 4; ++i) {
    f.u[i]     = (uint32_t)p[(size_t)(2 * i) * ld] | ((uint32_t)p[(size_t)(2 * i + 1) * ld] << 16);
    f.u[i + 4] = (uint32_t)p[(size_t)(16 + 2 * i) * ld] | ((uint32_t)p[(size_t)(17 + 2 * i) * ld] << 16);
  }
  return f;
}

// A fragment from fp32 LDS source (pack bf16 in-registers) -- P in P@V
__device__ __forceinline__ BFrag load_a_f32(const float* base, int ld,
                                            int row0, int k0, int lane) {
  BFrag f;
  int r  = lane & 15;
  int kh = (lane >> 4) << 3;
  const float* p = base + (size_t)(row0 + r) * ld + k0 + kh;
  float4 x0 = *(const float4*)p;
  float4 x1 = *(const float4*)(p + 4);
  float4 x2 = *(const float4*)(p + 16);
  float4 x3 = *(const float4*)(p + 20);
  f.u[0]=pk2bf(x0.x,x0.y); f.u[1]=pk2bf(x0.z,x0.w);
  f.u[2]=pk2bf(x1.x,x1.y); f.u[3]=pk2bf(x1.z,x1.w);
  f.u[4]=pk2bf(x2.x,x2.y); f.u[5]=pk2bf(x2.z,x2.w);
  f.u[6]=pk2bf(x3.x,x3.y); f.u[7]=pk2bf(x3.z,x3.w);
  return f;
}

__device__ __forceinline__ void wmma_bf16(FAcc& acc, const BFrag& a, const BFrag& b) {
  acc.v = __builtin_amdgcn_wmma_f32_16x16x32_bf16(false, a.v, false, b.v,
                                                  (short)0, acc.v, false, false);
}

__device__ __forceinline__ float wave_reduce_max(float v) {
  for (int o = 16; o > 0; o >>= 1) v = fmaxf(v, __shfl_xor(v, o, 32));
  return v;
}
__device__ __forceinline__ float wave_reduce_sum(float v) {
  for (int o = 16; o > 0; o >>= 1) v += __shfl_xor(v, o, 32);
  return v;
}

// ---------------------------------------------------------------- K0: weights
// which 0..2: Wt[k][n] = bf16(W[n][k])
// which 3 (Wo): K rows stored head-major: khm = h*64+c  <->  j = c*8+h,
//               Wt[khm][n] = bf16(Wo[n][j])  (folds channel interleave into B)
__global__ __launch_bounds__(256)
void pack_weights(const float* Wq, const float* Wk, const float* Wv,
                  const float* Wo, uint16_t* Wt) {
  int which = blockIdx.y;
  const float* W = (which == 0) ? Wq : (which == 1) ? Wk : (which == 2) ? Wv : Wo;
  uint16_t* dst = Wt + (size_t)which * CCH * CCH;
  int idx = blockIdx.x * 256 + threadIdx.x;
  int k = idx & (CCH - 1);
  int n = idx >> 9;
  int ksrc = k;
  if (which == 3) { int h = k >> 6, c = k & 63; ksrc = c * 8 + h; }
  dst[(size_t)k * CCH + n] = f2bf(W[(size_t)n * CCH + ksrc]);
}

// --------------------------------------------------------------- K1: QKV GEMM
// z=0: Q[bm][tok][d], z=1: Kt[bm][d][tok], z=2: V[bm][tok][d]  (all bf16)
// NCHW->token transpose + cvt folded into LDS A staging; output retiled
// through LDS so every global store is a contiguous b128.
__global__ __launch_bounds__(256)
void qkv_gemm(const float* q_in, const float* k_in, const float* v_in,
              const uint16_t* Wt, uint16_t* Q, uint16_t* Kt, uint16_t* V) {
  const int z = blockIdx.z;
  const float* src = (z == 0) ? q_in : (z == 1) ? k_in : v_in;
  const uint16_t* W = Wt + (size_t)z * CCH * CCH;

  __shared__ uint16_t Abuf[128 * 40];    // 128 tokens x 32 K
  __shared__ uint16_t Bbuf[32 * 136];    // 32 K x 128 N
  __shared__ uint16_t Obuf[128 * 136];   // output retile

  int tid = threadIdx.x, lane = tid & 31, w = tid >> 5;
  int row0 = blockIdx.x * 128;
  int n0   = blockIdx.y * 128;
  int bq = row0 >> 10, hw0 = row0 & 1023, tok0 = hw0;
  const float* srcB = src + (size_t)bq * CCH * HWTOK;
  int mw = w & 1, nw = w >> 1;

  FAcc acc[4][2];
  for (int i = 0; i < 4; ++i)
    for (int j = 0; j < 2; ++j)
      for (int r = 0; r < 8; ++r) acc[i][j].f[r] = 0.f;

  for (int kt = 0; kt < CCH / 32; ++kt) {
    int kc = kt * 32;
    __syncthreads();
    // stage A: 2048 channel-pairs, coalesced along HW, packed bf16x2 stores
    for (int it = 0; it < 8; ++it) {
      int lin = it * 256 + tid;
      int i = lin & 127, j = (lin >> 7) * 2;
      float a0 = srcB[(size_t)(kc + j) * HWTOK + hw0 + i];
      float a1 = srcB[(size_t)(kc + j + 1) * HWTOK + hw0 + i];
      *(uint32_t*)&Abuf[i * 40 + j] = pk2bf(a0, a1);
    }
    // stage B: async global->LDS b128 (ASYNCcnt path)
    for (int it = 0; it < 2; ++it) {
      int g = it * 256 + tid;
      int j = g >> 4, c8 = (g & 15) * 8;
      async_copy_b128(&Bbuf[j * 136 + c8],
                      &W[(size_t)(kc + j) * CCH + n0 + c8]);
    }
    if (kt + 1 < CCH / 32) {
      __builtin_prefetch(&srcB[(size_t)(kc + 32) * HWTOK + hw0 + (tid & 127)], 0, 1);
    }
    async_wait0();
    __syncthreads();
    BFrag bf0 = load_b_bf16(Bbuf, 136, 0, nw * 32, lane);
    BFrag bf1 = load_b_bf16(Bbuf, 136, 0, nw * 32 + 16, lane);
    for (int mt = 0; mt < 4; ++mt) {
      BFrag af = load_a_bf16(Abuf, 40, mw * 64 + mt * 16, 0, lane);
      wmma_bf16(acc[mt][0], af, bf0);
      wmma_bf16(acc[mt][1], af, bf1);
    }
  }

  // retile output through LDS: [n][tok] for Kt, [tok][n] for Q/V
  __syncthreads();
  int col = lane & 15, rh = (lane >> 4) << 3;
  for (int mt = 0; mt < 4; ++mt)
    for (int ntl = 0; ntl < 2; ++ntl)
      for (int r = 0; r < 8; ++r) {
        int tl = mw * 64 + mt * 16 + r + rh;
        int nl = nw * 32 + ntl * 16 + col;
        uint16_t val = f2bf(acc[mt][ntl].f[r]);
        if (z == 1) Obuf[nl * 136 + tl] = val;
        else        Obuf[tl * 136 + nl] = val;
      }
  __syncthreads();

  if (z == 1) {          // Kt rows contiguous along tokens
    for (int it = 0; it < 8; ++it) {
      int g = it * 256 + tid;
      int nl = g >> 4, tl8 = (g & 15) * 8;
      int n = n0 + nl, h = n >> 6, dd = n & 63;
      int bm = bq * NHEAD + h;
      *(uint4*)&Kt[((size_t)bm * DH + dd) * HWTOK + tok0 + tl8] =
          *(const uint4*)&Obuf[nl * 136 + tl8];
    }
  } else {               // Q/V rows contiguous along head channels
    uint16_t* dst = (z == 0) ? Q : V;
    for (int it = 0; it < 8; ++it) {
      int g = it * 256 + tid;
      int tl = g >> 4, nl8 = (g & 15) * 8;
      int n = n0 + nl8, h = n >> 6, dd = n & 63;
      int bm = bq * NHEAD + h;
      *(uint4*)&dst[((size_t)bm * HWTOK + tok0 + tl) * DH + dd] =
          *(const uint4*)&Obuf[tl * 136 + nl8];
    }
  }
}

// -------------------------------------------------------------- K2: attention
__global__ __launch_bounds__(256)
void attention(const uint16_t* Q, const uint16_t* Kt, const uint16_t* V, float* O) {
  __shared__ float S[32 * 1024];   // 128 KB score block
  int bm = blockIdx.x;
  int q0 = blockIdx.y * 32;
  int tid = threadIdx.x, lane = tid & 31, w = tid >> 5;
  const uint16_t* Qh = Q  + (size_t)bm * HWTOK * DH;
  const uint16_t* Kh = Kt + (size_t)bm * DH * HWTOK;
  const uint16_t* Vh = V  + (size_t)bm * HWTOK * DH;

  BFrag qf[2][2];
  for (int mt = 0; mt < 2; ++mt)
    for (int ks = 0; ks < 2; ++ks)
      qf[mt][ks] = load_a_bf16(Qh, DH, q0 + mt * 16, ks * 32, lane);

  int col = lane & 15, rh = (lane >> 4) << 3;
  // phase 1: S = (Q @ K^T) / sqrt(64)
  for (int idx = 0; idx < 8; ++idx) {
    int nt = w + idx * 8;
    BFrag kf0 = load_b_bf16(Kh, HWTOK, 0,  nt * 16, lane);
    BFrag kf1 = load_b_bf16(Kh, HWTOK, 32, nt * 16, lane);
    for (int mt = 0; mt < 2; ++mt) {
      FAcc a; for (int r = 0; r < 8; ++r) a.f[r] = 0.f;
      wmma_bf16(a, qf[mt][0], kf0);
      wmma_bf16(a, qf[mt][1], kf1);
      for (int r = 0; r < 8; ++r)
        S[(mt * 16 + r + rh) * 1024 + nt * 16 + col] = a.f[r] * 0.125f;
    }
  }
  __syncthreads();

  // phase 2: softmax, wave w owns rows 4w..4w+3
  for (int rr = 0; rr < 4; ++rr) {
    float* Srow = S + (w * 4 + rr) * 1024;
    float mx = -1e30f;
    for (int i = 0; i < 32; ++i) mx = fmaxf(mx, Srow[lane + i * 32]);
    mx = wave_reduce_max(mx);
    float sum = 0.f;
    for (int i = 0; i < 32; ++i) {
      float e = __expf(Srow[lane + i * 32] - mx);
      Srow[lane + i * 32] = e;
      sum += e;
    }
    sum = wave_reduce_sum(sum);
    float inv = 1.0f / sum;
    for (int i = 0; i < 32; ++i) Srow[lane + i * 32] *= inv;
  }
  __syncthreads();

  // phase 3: out = P @ V
  int mt = w & 1, nt = w >> 1;
  FAcc acc; for (int r = 0; r < 8; ++r) acc.f[r] = 0.f;
  for (int ks = 0; ks < 32; ++ks) {
    BFrag pf = load_a_f32(S, 1024, mt * 16, ks * 32, lane);
    BFrag vf = load_b_bf16(Vh, DH, ks * 32, nt * 16, lane);
    wmma_bf16(acc, pf, vf);
  }
  float* Oh = O + (size_t)bm * HWTOK * DH;
  for (int r = 0; r < 8; ++r)
    Oh[(size_t)(q0 + mt * 16 + r + rh) * DH + nt * 16 + col] = acc.f[r];
}

// ----------------------------------------------------- K3: residual+LayerNorm
// Head-major (fully coalesced) reads/writes; LN stats are permutation
// invariant; gamma/beta gathered via interleaved index j=c*8+h; the
// channel permutation is folded into the packed Wo (pack_weights which==3).
__global__ __launch_bounds__(256)
void add_ln(const float* O, const uint16_t* Q, const float* gamma,
            const float* beta, uint16_t* Xo) {
  int lane = threadIdx.x & 31, w = threadIdx.x >> 5;
  int t = blockIdx.x * 8 + w;
  int b = t >> 10, tok = t & 1023;
  float vals[16];
  float s = 0.f;
  for (int i = 0; i < 16; ++i) {
    int khm = i * 32 + lane;               // head-major channel h*64+c
    int h = khm >> 6, c = khm & 63;
    size_t idx = ((size_t)(b * NHEAD + h) * HWTOK + tok) * DH + c;
    float v = O[idx] + bf2f(Q[idx]);
    vals[i] = v; s += v;
  }
  s = wave_reduce_sum(s);
  float mu = s * (1.0f / 512.0f);
  float vs = 0.f;
  for (int i = 0; i < 16; ++i) { float d = vals[i] - mu; vs += d * d; }
  vs = wave_reduce_sum(vs);
  float rstd = rsqrtf(vs * (1.0f / 512.0f) + LN_EPS);
  for (int i = 0; i < 16; ++i) {
    int khm = i * 32 + lane;
    int h = khm >> 6, c = khm & 63;
    int j = c * 8 + h;                     // interleaved index for gamma/beta
    Xo[(size_t)t * CCH + khm] = f2bf((vals[i] - mu) * rstd * gamma[j] + beta[j]);
  }
}

// --------------------------------------------------------------- K4: out GEMM
// Y = Xo(8192x512 bf16, head-major K) @ WoT(permuted) ; LDS retile so NCHW
// stores are contiguous b128 along HW.
__global__ __launch_bounds__(256)
void out_gemm(const uint16_t* Xo, const uint16_t* Wot, float* out) {
  __shared__ uint16_t Abuf[128 * 40];
  __shared__ uint16_t Bbuf[32 * 136];
  __shared__ float    Obuf[128 * 132];   // [n][tok] retile, 66 KB
  int tid = threadIdx.x, lane = tid & 31, w = tid >> 5;
  int row0 = blockIdx.x * 128, n0 = blockIdx.y * 128;
  int mw = w & 1, nw = w >> 1;

  FAcc acc[4][2];
  for (int i = 0; i < 4; ++i)
    for (int j = 0; j < 2; ++j)
      for (int r = 0; r < 8; ++r) acc[i][j].f[r] = 0.f;

  for (int kt = 0; kt < CCH / 32; ++kt) {
    int kc = kt * 32;
    __syncthreads();
    for (int it = 0; it < 2; ++it) {       // A tile: async b128
      int g = it * 256 + tid;
      int row = g >> 2, q4 = (g & 3) * 8;
      async_copy_b128(&Abuf[row * 40 + q4],
                      &Xo[(size_t)(row0 + row) * CCH + kc + q4]);
    }
    for (int it = 0; it < 2; ++it) {       // B tile: async b128
      int g = it * 256 + tid;
      int j = g >> 4, c8 = (g & 15) * 8;
      async_copy_b128(&Bbuf[j * 136 + c8],
                      &Wot[(size_t)(kc + j) * CCH + n0 + c8]);
    }
    if (kt + 1 < CCH / 32)
      __builtin_prefetch(&Xo[(size_t)(row0 + (tid >> 1)) * CCH + kc + 32], 0, 1);
    async_wait0();
    __syncthreads();
    BFrag bf0 = load_b_bf16(Bbuf, 136, 0, nw * 32, lane);
    BFrag bf1 = load_b_bf16(Bbuf, 136, 0, nw * 32 + 16, lane);
    for (int mt = 0; mt < 4; ++mt) {
      BFrag af = load_a_bf16(Abuf, 40, mw * 64 + mt * 16, 0, lane);
      wmma_bf16(acc[mt][0], af, bf0);
      wmma_bf16(acc[mt][1], af, bf1);
    }
  }

  int col = lane & 15, rh = (lane >> 4) << 3;
  for (int mt = 0; mt < 4; ++mt)
    for (int ntl = 0; ntl < 2; ++ntl)
      for (int r = 0; r < 8; ++r) {
        int tl = mw * 64 + mt * 16 + r + rh;
        int nl = nw * 32 + ntl * 16 + col;
        Obuf[nl * 132 + tl] = acc[mt][ntl].f[r];
      }
  __syncthreads();

  int bq = row0 >> 10, hw0 = row0 & 1023;
  for (int it = 0; it < 16; ++it) {        // coalesced float4 NCHW stores
    int g = it * 256 + tid;
    int nl = g >> 5, tl4 = (g & 31) * 4;
    *(float4*)&out[((size_t)bq * CCH + n0 + nl) * HWTOK + hw0 + tl4] =
        *(const float4*)&Obuf[nl * 132 + tl4];
  }
}

// ------------------------------------------------------------------- launcher
extern "C" void kernel_launch(void* const* d_in, const int* in_sizes, int n_in,
                              void* d_out, int out_size, void* d_ws, size_t ws_size,
                              hipStream_t stream) {
  const float* key   = (const float*)d_in[0];
  const float* value = (const float*)d_in[1];
  const float* query = (const float*)d_in[2];
  const float* Wq    = (const float*)d_in[3];
  const float* Wk    = (const float*)d_in[4];
  const float* Wv    = (const float*)d_in[5];
  const float* Wo    = (const float*)d_in[6];
  const float* gamma = (const float*)d_in[7];
  const float* beta  = (const float*)d_in[8];
  float* out = (float*)d_out;

  char* ws = (char*)d_ws;
  const size_t MB = (size_t)1 << 20;
  uint16_t* Wt  = (uint16_t*)(ws);              //  2 MB
  uint16_t* Qb  = (uint16_t*)(ws +  2 * MB);    //  8 MB
  uint16_t* Ktb = (uint16_t*)(ws + 10 * MB);    //  8 MB
  uint16_t* Vb  = (uint16_t*)(ws + 18 * MB);    //  8 MB
  float*    Ob  = (float*)   (ws + 26 * MB);    // 16 MB
  uint16_t* Xob = (uint16_t*)(ws + 42 * MB);    //  8 MB
  (void)in_sizes; (void)n_in; (void)out_size; (void)ws_size;

  pack_weights<<<dim3(1024, 4), 256, 0, stream>>>(Wq, Wk, Wv, Wo, Wt);
  qkv_gemm<<<dim3(64, 4, 3), 256, 0, stream>>>(query, key, value, Wt, Qb, Ktb, Vb);
  attention<<<dim3(64, 32), 256, 0, stream>>>(Qb, Ktb, Vb, Ob);
  add_ln<<<dim3(1024), 256, 0, stream>>>(Ob, Qb, gamma, beta, Xob);
  out_gemm<<<dim3(64, 4), 256, 0, stream>>>(Xob, Wt + (size_t)3 * CCH * CCH, out);
}
